// Base_37993280700515
// MI455X (gfx1250) — matrix operand order repaired and checked by reading, not compiled, not run on hardware
//
#include <hip/hip_runtime.h>

typedef __attribute__((ext_vector_type(16))) _Float16 v16h;
typedef __attribute__((ext_vector_type(8)))  float    v8f;

#define N_NODES 200000
#define N_EDGES 600000
#define LAYERS  4
#define ND      64
#define ED      12

// ---------- helpers ----------

__device__ inline v8f wmma32(v16h a, v16h b, v8f c) {
  // D = A(16x32 f16) * B(32x16 f16) + C(16x16 f32)
  return __builtin_amdgcn_wmma_f32_16x16x32_f16(
      /*neg_a=*/false, a, /*neg_b=*/false, b,
      /*c_mod=*/(short)0, c, /*reuse_a=*/false, /*reuse_b=*/false);
}

// A-fragment (16x32 f16) for lane (m = lane&15, h = lane>>4) from a row-major
// f16 row pointer. Element j <-> k = kb*32 + (j>>3)*16 + h*8 + (j&7).
// Both 8-element groups are contiguous in k, so two 16B loads suffice.
__device__ inline v16h load_a_frag(const _Float16* __restrict__ row, int kb, int h) {
  const _Float16* p = row + kb * 32 + h * 8;
  v16h a;
  *(float4*)&a         = *(const float4*)p;         // k .. k+7
  *(((float4*)&a) + 1) = *(const float4*)(p + 16);  // k+16 .. k+23
  return a;
}

// order-preserving float<->uint encoding for atomicMax-based segment max
__device__ inline unsigned enc_f(float f) {
  unsigned u = __float_as_uint(f);
  return (u & 0x80000000u) ? ~u : (u | 0x80000000u);
}
__device__ inline float dec_f(unsigned u) {
  return (u & 0x80000000u) ? __uint_as_float(u ^ 0x80000000u)
                           : __uint_as_float(~u);
}

// ---------- kernels ----------

// x = atom_emb[atoms]; also f16 copy
__global__ void k_embed(const int* __restrict__ atoms,
                        const float* __restrict__ emb,
                        float* __restrict__ x, _Float16* __restrict__ xh,
                        int total) {
  int idx = blockIdx.x * blockDim.x + threadIdx.x;
  if (idx >= total) return;
  int i = idx >> 6, d = idx & 63;
  float v = emb[atoms[i] * ND + d];
  x[idx]  = v;
  xh[idx] = (_Float16)v;
}

// pack a row-major [K,64] f32 weight into f16 B fragments:
// frag = kb*4 + nb ; within frag: lane holds column n = nb*16 + (lane&15),
// k = kb*32 + (j>>3)*16 + (lane>>4)*8 + (j&7)  (j = 0..15), 16 f16 contiguous per lane.
__global__ void k_pack(const float* __restrict__ W, _Float16* __restrict__ out,
                       int kfrags) {
  int t = blockIdx.x * blockDim.x + threadIdx.x;
  int total = kfrags * 4 * 512;
  if (t >= total) return;
  int frag = t >> 9;
  int kb = frag >> 2, nb = frag & 3;
  int lane = (t >> 4) & 31, j = t & 15;
  int k = kb * 32 + (j >> 3) * 16 + (lane >> 4) * 8 + (j & 7);
  int n = nb * 16 + (lane & 15);
  out[t] = (_Float16)W[k * ND + n];
}

// q/k/v/skip projections: one wave per 16-row tile, 4 matrices x 4 N-tiles x 2 K-steps
__global__ void __launch_bounds__(256)
k_node_proj(const _Float16* __restrict__ xh, const _Float16* __restrict__ pw,
            const float* __restrict__ bq, const float* __restrict__ bk,
            const float* __restrict__ bv, const float* __restrict__ bs,
            float* __restrict__ oq, float* __restrict__ ok,
            float* __restrict__ ov, float* __restrict__ oo, int ntiles) {
  int wid = (blockIdx.x * blockDim.x + threadIdx.x) >> 5;
  if (wid >= ntiles) return;
  int lane = threadIdx.x & 31;
  int h = lane >> 4, nidx = lane & 15;
  const _Float16* xrow = xh + (size_t)(wid * 16 + (lane & 15)) * ND;
  v16h a0 = load_a_frag(xrow, 0, h);
  v16h a1 = load_a_frag(xrow, 1, h);
  const float* biases[4] = {bq, bk, bv, bs};
  float* outs[4] = {oq, ok, ov, oo};
#pragma unroll
  for (int m = 0; m < 4; ++m) {
    const _Float16* pm = pw + m * 4096;
    const float* bb = biases[m];
    float* om = outs[m];
#pragma unroll
    for (int nb = 0; nb < 4; ++nb) {
      float bval = bb[nb * 16 + nidx];
      v8f c;
#pragma unroll
      for (int r = 0; r < 8; ++r) c[r] = bval;
      v16h b0 = *(const v16h*)(pm + (0 * 4 + nb) * 512 + lane * 16);
      c = wmma32(a0, b0, c);
      v16h b1 = *(const v16h*)(pm + (1 * 4 + nb) * 512 + lane * 16);
      c = wmma32(a1, b1, c);
#pragma unroll
      for (int r = 0; r < 8; ++r)
        om[(size_t)(wid * 16 + r + 8 * h) * ND + nb * 16 + nidx] = c[r];
    }
  }
}

// e_proj = edge_emb[edge_ids] @ We   ([E,12] x [12,64])
__global__ void __launch_bounds__(256)
k_edge_proj(const int* __restrict__ edge_ids, const float* __restrict__ edge_emb,
            const float* __restrict__ We, float* __restrict__ eproj, int total) {
  __shared__ float sW[ED * ND];
  __shared__ float sE[3 * ED];
  for (int i = threadIdx.x; i < ED * ND; i += blockDim.x) sW[i] = We[i];
  for (int i = threadIdx.x; i < 3 * ED; i += blockDim.x) sE[i] = edge_emb[i];
  __syncthreads();
  int idx = blockIdx.x * blockDim.x + threadIdx.x;
  if (idx >= total) return;
  int e = idx >> 6, d = idx & 63;
  int id = edge_ids[e];
  float acc = 0.f;
#pragma unroll
  for (int j = 0; j < ED; ++j) acc += sE[id * ED + j] * sW[j * ND + d];
  eproj[idx] = acc;
}

__global__ void k_init_nodes(unsigned* __restrict__ amax, float* __restrict__ den,
                             int n) {
  int i = blockIdx.x * blockDim.x + threadIdx.x;
  if (i >= n) return;
  amax[i] = 0u;   // < enc(-inf): identity for segment-max
  den[i]  = 0.f;
}

// alpha[e] = dot(q[tgt], k[src]+eproj[e]) * (1/8); segment max via atomicMax
__global__ void __launch_bounds__(256)
k_alpha(const float* __restrict__ q, const float* __restrict__ kk,
        const float* __restrict__ eproj, const int* __restrict__ src,
        const int* __restrict__ tgt, float* __restrict__ alpha,
        unsigned* __restrict__ amax, int nE) {
  int e = blockIdx.x * blockDim.x + threadIdx.x;
  if (e >= nE) return;
  int s = src[e], t = tgt[e];
  const float4* q4 = (const float4*)(q + (size_t)t * ND);
  const float4* k4 = (const float4*)(kk + (size_t)s * ND);
  const float4* e4 = (const float4*)(eproj + (size_t)e * ND);
  float acc = 0.f;
#pragma unroll
  for (int j = 0; j < 16; ++j) {
    float4 qa = q4[j], ka = k4[j], ea = e4[j];
    acc += qa.x * (ka.x + ea.x) + qa.y * (ka.y + ea.y) +
           qa.z * (ka.z + ea.z) + qa.w * (ka.w + ea.w);
  }
  acc *= 0.125f;  // 1/sqrt(64)
  alpha[e] = acc;
  atomicMax(&amax[t], enc_f(acc));
}

// ex = exp(alpha - amax[tgt]); den[tgt] += ex  (ex overwrites alpha)
__global__ void __launch_bounds__(256)
k_exp(float* __restrict__ alpha, const unsigned* __restrict__ amax,
      const int* __restrict__ tgt, float* __restrict__ den, int nE) {
  int e = blockIdx.x * blockDim.x + threadIdx.x;
  if (e >= nE) return;
  int t = tgt[e];
  float ex = expf(alpha[e] - dec_f(amax[t]));
  alpha[e] = ex;
  atomicAdd(&den[t], ex);
}

// out[tgt] += (v[src]+eproj) * ex/(den[tgt]+1e-16)
__global__ void __launch_bounds__(256)
k_scatter(const float* __restrict__ alpha, const float* __restrict__ den,
          const float* __restrict__ v, const float* __restrict__ eproj,
          const int* __restrict__ src, const int* __restrict__ tgt,
          float* __restrict__ out, int total) {
  int idx = blockIdx.x * blockDim.x + threadIdx.x;
  if (idx >= total) return;
  int e = idx >> 6, d = idx & 63;
  int s = src[e], t = tgt[e];
  float w = alpha[e] / (den[t] + 1e-16f);
  atomicAdd(&out[(size_t)t * ND + d], (v[(size_t)s * ND + d] + eproj[idx]) * w);
}

// concat [r2c | c2r] -> f16 [N,128]
__global__ void k_concat(const float* __restrict__ o0, const float* __restrict__ o1,
                         _Float16* __restrict__ ch, int total) {
  int idx = blockIdx.x * blockDim.x + threadIdx.x;
  if (idx >= total) return;
  int i = idx >> 7, d = idx & 127;
  float val = (d < 64) ? o0[(size_t)i * 64 + d] : o1[(size_t)i * 64 + (d - 64)];
  ch[idx] = (_Float16)val;
}

// x = gelu(concat @ aggr_W + aggr_b), K=128 -> 4 WMMA k-steps; writes f32 + f16
__global__ void __launch_bounds__(256)
k_aggr(const _Float16* __restrict__ ch, const _Float16* __restrict__ pw,
       const float* __restrict__ bias, float* __restrict__ xo,
       _Float16* __restrict__ xho, int ntiles) {
  int wid = (blockIdx.x * blockDim.x + threadIdx.x) >> 5;
  if (wid >= ntiles) return;
  int lane = threadIdx.x & 31;
  int h = lane >> 4, nidx = lane & 15;
  const _Float16* arow = ch + (size_t)(wid * 16 + (lane & 15)) * 128;
  v16h a[4];
#pragma unroll
  for (int kb = 0; kb < 4; ++kb) a[kb] = load_a_frag(arow, kb, h);
#pragma unroll
  for (int nb = 0; nb < 4; ++nb) {
    float bval = bias[nb * 16 + nidx];
    v8f c;
#pragma unroll
    for (int r = 0; r < 8; ++r) c[r] = bval;
#pragma unroll
    for (int kb = 0; kb < 4; ++kb) {
      v16h b = *(const v16h*)(pw + (kb * 4 + nb) * 512 + lane * 16);
      c = wmma32(a[kb], b, c);
    }
#pragma unroll
    for (int r = 0; r < 8; ++r) {
      float hv = c[r];
      float g = 0.5f * hv * (1.0f + erff(hv * 0.70710678118654752f));
      size_t off = (size_t)(wid * 16 + r + 8 * h) * ND + nb * 16 + nidx;
      xo[off]  = g;
      xho[off] = (_Float16)g;
    }
  }
}

// ---------- launch ----------

extern "C" void kernel_launch(void* const* d_in, const int* in_sizes, int n_in,
                              void* d_out, int out_size, void* d_ws, size_t ws_size,
                              hipStream_t stream) {
  const int*   atoms      = (const int*)d_in[0];
  const int*   edge_index = (const int*)d_in[1];
  const int*   edge_ids   = (const int*)d_in[2];
  const float* atom_emb   = (const float*)d_in[3];
  const float* edge_emb   = (const float*)d_in[4];
  const float* Wq = (const float*)d_in[5];
  const float* bq = (const float*)d_in[6];
  const float* Wk = (const float*)d_in[7];
  const float* bk = (const float*)d_in[8];
  const float* Wv = (const float*)d_in[9];
  const float* bv = (const float*)d_in[10];
  const float* We = (const float*)d_in[11];
  const float* Ws = (const float*)d_in[12];
  const float* bs = (const float*)d_in[13];
  const float* aggr_W = (const float*)d_in[14];
  const float* aggr_b = (const float*)d_in[15];

  size_t off = 0;
  auto carve = [&](size_t bytes) -> char* {
    off = (off + 255) & ~(size_t)255;
    char* p = (char*)d_ws + off;
    off += bytes;
    return p;
  };

  float*    x     = (float*)   carve((size_t)N_NODES * ND * 4);
  _Float16* xh    = (_Float16*)carve((size_t)N_NODES * ND * 2);
  float*    q     = (float*)   carve((size_t)N_NODES * ND * 4);
  float*    kbuf  = (float*)   carve((size_t)N_NODES * ND * 4);
  float*    vbuf  = (float*)   carve((size_t)N_NODES * ND * 4);
  float*    o0    = (float*)   carve((size_t)N_NODES * ND * 4);
  float*    o1    = (float*)   carve((size_t)N_NODES * ND * 4);
  float*    eproj = (float*)   carve((size_t)N_EDGES * ND * 4);
  _Float16* cath  = (_Float16*)carve((size_t)N_NODES * 128 * 2);
  float*    alpha = (float*)   carve((size_t)N_EDGES * 4);
  unsigned* amax  = (unsigned*)carve((size_t)N_NODES * 4);
  float*    den   = (float*)   carve((size_t)N_NODES * 4);
  _Float16* npw   = (_Float16*)carve((size_t)8 * 4 * 4096 * 2);  // [cfg][mat] packed
  _Float16* apw   = (_Float16*)carve((size_t)8192 * 2);          // aggr packed

  const int ntiles = N_NODES / 16;                 // 12500
  const int gemm_blocks = (ntiles * 32 + 255) / 256;

  // embedding
  k_embed<<<(N_NODES * ND) / 256, 256, 0, stream>>>(atoms, atom_emb, x, xh,
                                                    N_NODES * ND);

  // pack all per-(dir,layer) node weights [Wq,Wk,Wv,Ws] and the aggregator
  for (int cfg = 0; cfg < 2 * LAYERS; ++cfg) {
    const float* mats[4] = {Wq + (size_t)cfg * ND * ND, Wk + (size_t)cfg * ND * ND,
                            Wv + (size_t)cfg * ND * ND, Ws + (size_t)cfg * ND * ND};
    for (int m = 0; m < 4; ++m)
      k_pack<<<(2 * 4 * 512 + 255) / 256, 256, 0, stream>>>(
          mats[m], npw + ((size_t)cfg * 4 + m) * 4096, 2);
  }
  k_pack<<<(4 * 4 * 512 + 255) / 256, 256, 0, stream>>>(aggr_W, apw, 4);

  const int* ei0 = edge_index;            // row 0: src (r2c)
  const int* ei1 = edge_index + N_EDGES;  // row 1: tgt (r2c)

  for (int hop = 0; hop < LAYERS; ++hop) {
    for (int dir = 0; dir < 2; ++dir) {
      const int cfg = dir * LAYERS + hop;
      const int* src = (dir == 0) ? ei0 : ei1;
      const int* tgt = (dir == 0) ? ei1 : ei0;
      float* outbuf = (dir == 0) ? o0 : o1;

      k_node_proj<<<gemm_blocks, 256, 0, stream>>>(
          xh, npw + (size_t)cfg * 4 * 4096,
          bq + (size_t)cfg * ND, bk + (size_t)cfg * ND,
          bv + (size_t)cfg * ND, bs + (size_t)cfg * ND,
          q, kbuf, vbuf, outbuf, ntiles);

      k_edge_proj<<<(N_EDGES * ND) / 256, 256, 0, stream>>>(
          edge_ids, edge_emb, We + (size_t)cfg * ED * ND, eproj, N_EDGES * ND);

      k_init_nodes<<<(N_NODES + 255) / 256, 256, 0, stream>>>(amax, den, N_NODES);

      k_alpha<<<(N_EDGES + 255) / 256, 256, 0, stream>>>(
          q, kbuf, eproj, src, tgt, alpha, amax, N_EDGES);

      k_exp<<<(N_EDGES + 255) / 256, 256, 0, stream>>>(alpha, amax, tgt, den,
                                                       N_EDGES);

      k_scatter<<<(N_EDGES * ND) / 256, 256, 0, stream>>>(
          alpha, den, vbuf, eproj, src, tgt, outbuf, N_EDGES * ND);
    }

    k_concat<<<(N_NODES * 128) / 256, 256, 0, stream>>>(o0, o1, cath,
                                                        N_NODES * 128);

    float* xout = (hop == LAYERS - 1) ? (float*)d_out : x;
    k_aggr<<<gemm_blocks, 256, 0, stream>>>(cath, apw, aggr_b, xout, xh, ntiles);
  }
}
